// HopfieldRouter_16776142258852
// MI455X (gfx1250) — matrix operand order repaired and checked by reading, not compiled, not run on hardware
//
#include <hip/hip_runtime.h>
#include <hip/hip_bf16.h>

typedef __attribute__((ext_vector_type(16))) _Float16 v16h;
typedef __attribute__((ext_vector_type(8)))  _Float16 v8h;
typedef __attribute__((ext_vector_type(8)))  float    v8f;
typedef __attribute__((ext_vector_type(4)))  int      v4i;

#define LN_EPS   1e-5f
#define NHEADS   4
#define ADIM     256
#define HADIM    1024     // NHEADS*ADIM
#define DMODEL   768
#define BATCH    32
#define NMEM     100000
#define SCALE_B  2.0f
#define NROWS    128      // NHEADS*BATCH logits rows
#define NEG_INF  (-3.402823466e38f)
#define SLDA     784      // 768 + 16 pad (halves): 16B aligned rows, spread LDS banks
#define ASPLITS  25       // n-splits for attn apply: 100000 = 25 * 4000
#define SCHUNK   16       // chunks per logits row for softmax stats: 100000/16 = 6250

// gfx1250 async global->LDS path (ASYNCcnt). Signature confirmed by compiler probe:
// (v4i addrspace(1)*, v4i addrspace(3)*, imm offset, imm cpol)
#if defined(__has_builtin)
# if __has_builtin(__builtin_amdgcn_global_load_async_to_lds_b128) && \
     __has_builtin(__builtin_amdgcn_s_wait_asynccnt)
#  define HAVE_ASYNC_LDS 1
# endif
#endif
#ifndef HAVE_ASYNC_LDS
# define HAVE_ASYNC_LDS 0
#endif

// ---------------- WMMA fragment helpers (wave32, 16x16x32 f16) ----------------
// A (16x32, row-major src, ld in halves): M=lane%16, K=16*(lane/16)+0..15 (contiguous)
static __device__ __forceinline__ v16h load_a_frag_f16(const _Float16* base, int ld) {
    int lane = threadIdx.x & 31;
    int m = lane & 15, s = lane >> 4;
    const _Float16* p = base + (size_t)m * ld + s * 16;
    v8h lo = *reinterpret_cast<const v8h*>(p);
    v8h hi = *reinterpret_cast<const v8h*>(p + 8);
    v16h a;
#pragma unroll
    for (int i = 0; i < 8; ++i) { a[i] = lo[i]; a[8 + i] = hi[i]; }
    return a;
}

// B (32x16) from an "N-major" f16 source: src[(n)*ld + k] — contiguous K per lane
static __device__ __forceinline__ v16h load_b_frag_f16_nmajor(const _Float16* base, int ld) {
    int lane = threadIdx.x & 31;
    int n = lane & 15, s = lane >> 4;
    const _Float16* p = base + (size_t)n * ld + s * 16;
    v8h lo = *reinterpret_cast<const v8h*>(p);
    v8h hi = *reinterpret_cast<const v8h*>(p + 8);
    v16h b;
#pragma unroll
    for (int i = 0; i < 8; ++i) { b[i] = lo[i]; b[8 + i] = hi[i]; }
    return b;
}

// B (32x16) from an "N-major" f32 source (convert on the fly, 16B vector loads)
static __device__ __forceinline__ v16h load_b_frag_f32_nmajor(const float* base, int ld) {
    int lane = threadIdx.x & 31;
    int n = lane & 15, s = lane >> 4;
    const float* p = base + (size_t)n * ld + s * 16;
    v16h b;
#pragma unroll
    for (int i = 0; i < 16; i += 4) {
        float4 f = *reinterpret_cast<const float4*>(p + i);
        b[i] = (_Float16)f.x; b[i + 1] = (_Float16)f.y;
        b[i + 2] = (_Float16)f.z; b[i + 3] = (_Float16)f.w;
    }
    return b;
}

// B (32x16) from a "K-major" f16 source (transposed read, used on LDS tiles)
static __device__ __forceinline__ v16h load_b_frag_f16_kmajor(const _Float16* base, int ld) {
    int lane = threadIdx.x & 31;
    int n = lane & 15, s = lane >> 4;
    v16h b;
#pragma unroll
    for (int i = 0; i < 16; ++i) b[i] = base[(size_t)(s * 16 + i) * ld + n];
    return b;
}

#define WMMA_F16(a, b, c) \
    __builtin_amdgcn_wmma_f32_16x16x32_f16(false, (a), false, (b), (short)0, (c), false, false)

// ---------------- block reductions ----------------
static __device__ __forceinline__ float block_reduce_sum(float v, float* sbuf) {
    int t = threadIdx.x;
#pragma unroll
    for (int m = 16; m >= 1; m >>= 1) v += __shfl_xor(v, m);
    if ((t & 31) == 0) sbuf[t >> 5] = v;
    __syncthreads();
    int nw = blockDim.x >> 5;
    float r = (t < nw) ? sbuf[t] : 0.0f;
    if (t < 32) {
#pragma unroll
        for (int m = 16; m >= 1; m >>= 1) r += __shfl_xor(r, m);
        if (t == 0) sbuf[0] = r;
    }
    __syncthreads();
    r = sbuf[0];
    __syncthreads();
    return r;
}

static __device__ __forceinline__ float block_reduce_max(float v, float* sbuf) {
    int t = threadIdx.x;
#pragma unroll
    for (int m = 16; m >= 1; m >>= 1) v = fmaxf(v, __shfl_xor(v, m));
    if ((t & 31) == 0) sbuf[t >> 5] = v;
    __syncthreads();
    int nw = blockDim.x >> 5;
    float r = (t < nw) ? sbuf[t] : NEG_INF;
    if (t < 32) {
#pragma unroll
        for (int m = 16; m >= 1; m >>= 1) r = fmaxf(r, __shfl_xor(r, m));
        if (t == 0) sbuf[0] = r;
    }
    __syncthreads();
    r = sbuf[0];
    __syncthreads();
    return r;
}

// ---------------- K1: LN(query) -> xi0 = 2*(qn @ Wq^T + bq), f16 ----------------
__global__ __launch_bounds__(256) void qproj_kernel(
    const float* __restrict__ query, const float* __restrict__ g, const float* __restrict__ b,
    const float* __restrict__ Wq, const float* __restrict__ bq, _Float16* __restrict__ xi16) {
    __shared__ __align__(16) float sx[DMODEL];
    __shared__ float red[32];
    int bi = blockIdx.x, t = threadIdx.x;
    const float* row = query + (size_t)bi * DMODEL;
    float s = 0.f, ss = 0.f;
    for (int i = t; i < DMODEL; i += 256) { float v = row[i]; s += v; ss += v * v; sx[i] = v; }
    float S = block_reduce_sum(s, red);
    float SS = block_reduce_sum(ss, red);
    float mu = S / DMODEL;
    float var = SS / DMODEL - mu * mu;
    float rstd = rsqrtf(var + LN_EPS);
    __syncthreads();
    for (int i = t; i < DMODEL; i += 256) sx[i] = (sx[i] - mu) * rstd * g[i] + b[i];
    __syncthreads();
    for (int o = t; o < HADIM; o += 256) {
        const float* w = Wq + (size_t)o * DMODEL;
        float acc = 0.f;
        for (int i = 0; i < DMODEL; ++i) acc += sx[i] * w[i];
        xi16[(size_t)bi * HADIM + o] = (_Float16)(SCALE_B * (acc + bq[o]));
    }
}

// ---------------- K2: LN(mem) twice -> k = sn@Wk^T, v = pn@Wv^T (f16), + row L2 norms --
__global__ __launch_bounds__(256) void kvproj_kernel(
    const float* __restrict__ mem,
    const float* __restrict__ gs, const float* __restrict__ bs,
    const float* __restrict__ gp, const float* __restrict__ bp,
    const float* __restrict__ Wk, const float* __restrict__ bk,
    const float* __restrict__ Wv, const float* __restrict__ bv,
    _Float16* __restrict__ kbuf, _Float16* __restrict__ vbuf,
    float* __restrict__ mbnorm) {
    __shared__ __align__(16) _Float16 sS[16 * SLDA];
    __shared__ __align__(16) _Float16 sP[16 * SLDA];
    int t = threadIdx.x;
    int row0 = blockIdx.x * 16;
    int r = t >> 4, li = t & 15;           // 16 threads per memory row
    const float* mrow = mem + (size_t)(row0 + r) * DMODEL;

    float s = 0.f, ss = 0.f;
    for (int c = li; c < DMODEL; c += 16) { float x = mrow[c]; s += x; ss += x * x; }
#pragma unroll
    for (int m = 8; m >= 1; m >>= 1) { s += __shfl_xor(s, m); ss += __shfl_xor(ss, m); }
    float mu = s / DMODEL;
    float var = ss / DMODEL - mu * mu;
    float rstd = rsqrtf(var + LN_EPS);
    if (li == 0) mbnorm[row0 + r] = sqrtf(ss);   // raw L2 norm of the row
    for (int c = li; c < DMODEL; c += 16) {
        float nv = (mrow[c] - mu) * rstd;
        sS[r * SLDA + c] = (_Float16)(nv * gs[c] + bs[c]);
        sP[r * SLDA + c] = (_Float16)(nv * gp[c] + bp[c]);
    }
    __syncthreads();

    int w = t >> 5;                 // 8 waves, 128 output cols each
    int cb = w * 128;
    int lane = t & 31, nl = lane & 15, sg = lane >> 4;
#pragma unroll
    for (int half = 0; half < 2; ++half) {
        const _Float16* sA = half ? sP : sS;
        const float* W = half ? Wv : Wk;
        const float* bias = half ? bv : bk;
        _Float16* out = half ? vbuf : kbuf;
        for (int nt = 0; nt < 8; ++nt) {
            int cb2 = cb + nt * 16;
            v8f acc = {};
            for (int kk = 0; kk < DMODEL; kk += 32) {
                v16h a = load_a_frag_f16(sA + kk, SLDA);
                v16h bf = load_b_frag_f32_nmajor(W + (size_t)cb2 * DMODEL + kk, DMODEL);
                acc = WMMA_F16(a, bf, acc);
            }
            float bb = bias[cb2 + nl];
#pragma unroll
            for (int rr = 0; rr < 8; ++rr) {
                int mrow2 = sg * 8 + rr;
                out[(size_t)(row0 + mrow2) * HADIM + cb2 + nl] = (_Float16)(acc[rr] + bb);
            }
        }
    }
}

// ---------------- K3: logits[h*32+b, n] = xi[b,h,:] . k[n,h,:] ----------------
__global__ __launch_bounds__(128) void logits_kernel(
    const _Float16* __restrict__ xi16, const _Float16* __restrict__ kbuf,
    float* __restrict__ logits) {
    int h = blockIdx.y;
    int w = threadIdx.x >> 5;
    int nb = blockIdx.x * 64 + w * 16;       // 16 memory rows per wave
    if (nb >= NMEM) return;                  // wave-uniform guard
    const _Float16* kb = kbuf + (size_t)nb * HADIM + h * ADIM;
    const _Float16* xb = xi16 + h * ADIM;
    v8f acc0 = {}, acc1 = {};
    for (int kk = 0; kk < ADIM; kk += 32) {
        v16h bf = load_b_frag_f16_nmajor(kb + kk, HADIM);
        v16h a0 = load_a_frag_f16(xb + kk, HADIM);
        v16h a1 = load_a_frag_f16(xb + (size_t)16 * HADIM + kk, HADIM);
        acc0 = WMMA_F16(a0, bf, acc0);
        acc1 = WMMA_F16(a1, bf, acc1);
    }
    int lane = threadIdx.x & 31, nl = lane & 15, sg = lane >> 4;
#pragma unroll
    for (int r = 0; r < 8; ++r) {
        logits[(size_t)(h * 32 + sg * 8 + r) * NMEM + nb + nl] = acc0[r];
        logits[(size_t)(h * 32 + 16 + sg * 8 + r) * NMEM + nb + nl] = acc1[r];
    }
}

// ---------------- K4a: partial softmax stats per (row, chunk) ----------------
__global__ __launch_bounds__(256) void stats_partial_kernel(
    const float* __restrict__ logits, float* __restrict__ pmax, float* __restrict__ psum) {
    __shared__ float red[32];
    const int CH = NMEM / SCHUNK;   // 6250
    int row = blockIdx.y, c = blockIdx.x, t = threadIdx.x;
    const float* l = logits + (size_t)row * NMEM + (size_t)c * CH;
    float mx = NEG_INF;
    for (int i = t; i < CH; i += 256) mx = fmaxf(mx, l[i]);
    mx = block_reduce_max(mx, red);
    float se = 0.f;
    for (int i = t; i < CH; i += 256) se += __expf(l[i] - mx);
    se = block_reduce_sum(se, red);
    if (t == 0) { pmax[row * SCHUNK + c] = mx; psum[row * SCHUNK + c] = se; }
}

// ---------------- K4b: combine chunk stats (fixed order => deterministic) ------
__global__ __launch_bounds__(128) void stats_combine_kernel(
    const float* __restrict__ pmax, const float* __restrict__ psum, float* __restrict__ stats) {
    int row = threadIdx.x;
    if (row >= NROWS) return;
    float M = NEG_INF;
#pragma unroll
    for (int c = 0; c < SCHUNK; ++c) M = fmaxf(M, pmax[row * SCHUNK + c]);
    float S = 0.f;
#pragma unroll
    for (int c = 0; c < SCHUNK; ++c) S += psum[row * SCHUNK + c] * __expf(pmax[row * SCHUNK + c] - M);
    stats[2 * row] = M;
    stats[2 * row + 1] = 1.0f / S;
}

// ---------------- K4c: probs (f16) ----------------
__global__ __launch_bounds__(256) void probs_kernel(
    const float* __restrict__ logits, const float* __restrict__ stats,
    _Float16* __restrict__ p16) {
    int row = blockIdx.y;
    int i = blockIdx.x * 256 + threadIdx.x;
    if (i < NMEM) {
        float mx = stats[2 * row], inv = stats[2 * row + 1];
        p16[(size_t)row * NMEM + i] = (_Float16)(__expf(logits[(size_t)row * NMEM + i] - mx) * inv);
    }
}

// ------- K5a: partial[z][b, h*A+a] = sum_{n in split z} p[b,h,n] * kv[n, h*A+a] -------
__global__ __launch_bounds__(128) void attn_partial_kernel(
    const _Float16* __restrict__ p16, const _Float16* __restrict__ kv,
    float* __restrict__ partial) {
    __shared__ __align__(16) _Float16 sK[32 * 40];  // 32 n-rows x 32 a-cols, stride 40
    const int SPAN = NMEM / ASPLITS;                // 4000 rows, 125 tiles of 32
    int ab = blockIdx.x;            // 8 a-blocks of 32
    int h = blockIdx.y;             // 4 heads
    int z = blockIdx.z;             // 25 n-splits
    int t = threadIdx.x;
    int w = t >> 5;
    int m_tile = w & 1, a_tile = w >> 1;
    int ha0 = h * ADIM + ab * 32;
    const _Float16* prow = p16 + (size_t)(h * 32 + m_tile * 16) * NMEM;
    int sr = t >> 2, sc = (t & 3) * 8;   // cooperative stage: one 16B chunk per thread
    int n_begin = z * SPAN;
    v8f acc = {};
    for (int i = 0; i < SPAN; i += 32) {
        int n0 = n_begin + i;
        __syncthreads();   // previous tile fully consumed
        const _Float16* gsrc = kv + (size_t)(n0 + sr) * HADIM + ha0 + sc;
        _Float16* ldst = &sK[sr * 40 + sc];
#if HAVE_ASYNC_LDS
        __builtin_amdgcn_global_load_async_to_lds_b128(
            (__attribute__((address_space(1))) v4i*)gsrc,
            (__attribute__((address_space(3))) v4i*)ldst, 0, 0);
        __builtin_amdgcn_s_wait_asynccnt(0);
#else
        *reinterpret_cast<v8h*>(ldst) = *reinterpret_cast<const v8h*>(gsrc);
#endif
        __syncthreads();
        v16h a = load_a_frag_f16(prow + n0, NMEM);
        v16h bf = load_b_frag_f16_kmajor(sK + a_tile * 16, 40);
        acc = WMMA_F16(a, bf, acc);
    }
    int lane = t & 31, nl = lane & 15, sg = lane >> 4;
    int col = ha0 + a_tile * 16 + nl;
#pragma unroll
    for (int r = 0; r < 8; ++r) {
        int brow = m_tile * 16 + sg * 8 + r;
        partial[((size_t)z * BATCH + brow) * HADIM + col] = acc[r];
    }
}

// ------- K5b: deterministic fixed-order reduction over splits -------
__global__ __launch_bounds__(256) void attn_reduce_kernel(
    const float* __restrict__ partial, float scale, int write_f16,
    _Float16* __restrict__ dst16, float* __restrict__ dst32) {
    int idx = blockIdx.x * 256 + threadIdx.x;
    if (idx >= BATCH * HADIM) return;
    float s = 0.f;
#pragma unroll
    for (int z = 0; z < ASPLITS; ++z) s += partial[(size_t)z * (BATCH * HADIM) + idx];
    s *= scale;
    if (write_f16) dst16[idx] = (_Float16)s;
    else           dst32[idx] = s;
}

// ---------------- K7: out = hid @ Wo^T + bo, then L2 normalize (f16 out) --------
__global__ __launch_bounds__(256) void outproj_kernel(
    const float* __restrict__ hid, const float* __restrict__ Wo, const float* __restrict__ bo,
    _Float16* __restrict__ outn16) {
    __shared__ __align__(16) float sh[HADIM];
    __shared__ __align__(16) float so[DMODEL];
    __shared__ float red[32];
    int b = blockIdx.x, t = threadIdx.x;
    for (int i = t; i < HADIM; i += 256) sh[i] = hid[(size_t)b * HADIM + i];
    __syncthreads();
    for (int d = t; d < DMODEL; d += 256) {
        const float* w = Wo + (size_t)d * HADIM;
        float acc = bo[d];
        for (int i = 0; i < HADIM; ++i) acc += sh[i] * w[i];
        so[d] = acc;
    }
    __syncthreads();
    float ss = 0.f;
    for (int d = t; d < DMODEL; d += 256) ss += so[d] * so[d];
    ss = block_reduce_sum(ss, red);
    float inv = 1.0f / fmaxf(sqrtf(ss), 1e-12f);
    for (int d = t; d < DMODEL; d += 256) outn16[(size_t)b * DMODEL + d] = (_Float16)(so[d] * inv);
}

// ---------------- K8: scores[b,n] = outn . mem[n]/max(||mem[n]||,1e-12) --------
__global__ __launch_bounds__(128) void scores_kernel(
    const _Float16* __restrict__ outn16, const float* __restrict__ mem,
    const float* __restrict__ mbnorm, float* __restrict__ scores) {
    int n0 = blockIdx.x * 32;
    int w = threadIdx.x >> 5;
    int m_tile = w & 1, n_sub = w >> 1;
    int nb = n0 + n_sub * 16;
    v8f acc = {};
    for (int kk = 0; kk < DMODEL; kk += 32) {
        v16h a = load_a_frag_f16(outn16 + (size_t)(m_tile * 16) * DMODEL + kk, DMODEL);
        v16h bf = load_b_frag_f32_nmajor(mem + (size_t)nb * DMODEL + kk, DMODEL);
        acc = WMMA_F16(a, bf, acc);
    }
    int lane = threadIdx.x & 31, nl = lane & 15, sg = lane >> 4;
    int col = nb + nl;
    float inv = 1.0f / fmaxf(mbnorm[col], 1e-12f);
#pragma unroll
    for (int r = 0; r < 8; ++r)
        scores[(size_t)(m_tile * 16 + sg * 8 + r) * NMEM + col] = acc[r] * inv;
}

// ---------------- K9: top-3 per batch row (lowest index wins ties) --------------
__global__ __launch_bounds__(256) void topk_kernel(
    const float* __restrict__ scores, int* __restrict__ topi) {
    __shared__ float sv[256 * 3];
    __shared__ int si[256 * 3];
    int b = blockIdx.x, t = threadIdx.x;
    const float* row = scores + (size_t)b * NMEM;
    float v0 = NEG_INF, v1 = NEG_INF, v2 = NEG_INF;
    int i0 = 0x7fffffff, i1 = 0x7fffffff, i2 = 0x7fffffff;
    auto better = [](float v, int i, float bv, int bi) {
        return (v > bv) || (v == bv && i < bi);
    };
    for (int i = t; i < NMEM; i += 256) {
        float v = row[i];
        if (better(v, i, v0, i0)) { v2 = v1; i2 = i1; v1 = v0; i1 = i0; v0 = v; i0 = i; }
        else if (better(v, i, v1, i1)) { v2 = v1; i2 = i1; v1 = v; i1 = i; }
        else if (better(v, i, v2, i2)) { v2 = v; i2 = i; }
    }
    sv[t * 3 + 0] = v0; si[t * 3 + 0] = i0;
    sv[t * 3 + 1] = v1; si[t * 3 + 1] = i1;
    sv[t * 3 + 2] = v2; si[t * 3 + 2] = i2;
    __syncthreads();
    if (t == 0) {
        float b0 = NEG_INF, b1 = NEG_INF, b2 = NEG_INF;
        int j0 = 0x7fffffff, j1 = 0x7fffffff, j2 = 0x7fffffff;
        for (int c = 0; c < 256 * 3; ++c) {
            float v = sv[c]; int i = si[c];
            if (better(v, i, b0, j0)) { b2 = b1; j2 = j1; b1 = b0; j1 = j0; b0 = v; j0 = i; }
            else if (better(v, i, b1, j1)) { b2 = b1; j2 = j1; b1 = v; j1 = i; }
            else if (better(v, i, b2, j2)) { b2 = v; j2 = i; }
        }
        topi[b * 3 + 0] = j0; topi[b * 3 + 1] = j1; topi[b * 3 + 2] = j2;
    }
}

// ---------------- host launcher ----------------
extern "C" void kernel_launch(void* const* d_in, const int* in_sizes, int n_in,
                              void* d_out, int out_size, void* d_ws, size_t ws_size,
                              hipStream_t stream) {
    const float* query = (const float*)d_in[0];
    const float* mem   = (const float*)d_in[1];
    const float* Wq = (const float*)d_in[2];  const float* bq = (const float*)d_in[3];
    const float* Wk = (const float*)d_in[4];  const float* bk = (const float*)d_in[5];
    const float* Wv = (const float*)d_in[6];  const float* bv = (const float*)d_in[7];
    const float* Wo = (const float*)d_in[8];  const float* bo = (const float*)d_in[9];
    const float* g_state  = (const float*)d_in[10]; const float* b_state  = (const float*)d_in[11];
    const float* g_stored = (const float*)d_in[12]; const float* b_stored = (const float*)d_in[13];
    const float* g_proj   = (const float*)d_in[14]; const float* b_proj   = (const float*)d_in[15];

    int* topi = (int*)d_out;
    float* scores = (float*)d_out + BATCH * 3;   // outputs concatenated: indices then scores

    char* ws = (char*)d_ws;
    size_t off = 0;
    auto take = [&](size_t bytes) -> void* {
        void* p = ws + off;
        off = (off + bytes + 255) & ~(size_t)255;
        return p;
    };
    _Float16* kbuf    = (_Float16*)take((size_t)NMEM * HADIM * 2);
    _Float16* vbuf    = (_Float16*)take((size_t)NMEM * HADIM * 2);
    float*    logits  = (float*)   take((size_t)NROWS * NMEM * 4);
    _Float16* p16     = (_Float16*)take((size_t)NROWS * NMEM * 2);
    float*    stats   = (float*)   take((size_t)NROWS * 2 * 4);
    float*    pmax    = (float*)   take((size_t)NROWS * SCHUNK * 4);
    float*    psum    = (float*)   take((size_t)NROWS * SCHUNK * 4);
    float*    partial = (float*)   take((size_t)ASPLITS * BATCH * HADIM * 4);
    float*    mbnorm  = (float*)   take((size_t)NMEM * 4);
    _Float16* xi16    = (_Float16*)take((size_t)BATCH * HADIM * 2);
    float*    hid     = (float*)   take((size_t)BATCH * HADIM * 4);
    _Float16* outn16  = (_Float16*)take((size_t)BATCH * DMODEL * 2);

    qproj_kernel<<<BATCH, 256, 0, stream>>>(query, g_state, b_state, Wq, bq, xi16);
    kvproj_kernel<<<NMEM / 16, 256, 0, stream>>>(mem, g_stored, b_stored, g_proj, b_proj,
                                                 Wk, bk, Wv, bv, kbuf, vbuf, mbnorm);

    dim3 lgrid((NMEM + 63) / 64, NHEADS);
    dim3 sgrid(SCHUNK, NROWS);
    dim3 pgrid((NMEM + 255) / 256, NROWS);
    dim3 agrid(8, NHEADS, ASPLITS);
    const int rgrid = (BATCH * HADIM) / 256;   // 128
    for (int step = 0; step < 5; ++step) {
        logits_kernel<<<lgrid, 128, 0, stream>>>(xi16, kbuf, logits);
        stats_partial_kernel<<<sgrid, 256, 0, stream>>>(logits, pmax, psum);
        stats_combine_kernel<<<1, 128, 0, stream>>>(pmax, psum, stats);
        probs_kernel<<<pgrid, 256, 0, stream>>>(logits, stats, p16);
        if (step < 4) {
            attn_partial_kernel<<<agrid, 128, 0, stream>>>(p16, kbuf, partial);
            attn_reduce_kernel<<<rgrid, 256, 0, stream>>>(partial, SCALE_B, 1, xi16, nullptr);
        }
    }
    attn_partial_kernel<<<agrid, 128, 0, stream>>>(p16, vbuf, partial);
    attn_reduce_kernel<<<rgrid, 256, 0, stream>>>(partial, 1.0f, 0, nullptr, hid);
    outproj_kernel<<<BATCH, 256, 0, stream>>>(hid, Wo, bo, outn16);
    scores_kernel<<<NMEM / 32, 128, 0, stream>>>(outn16, mem, mbnorm, scores);
    topk_kernel<<<BATCH, 256, 0, stream>>>(scores, topi);
}
